// TransformerBlock_59072980189967
// MI455X (gfx1250) — matrix operand order repaired and checked by reading, not compiled, not run on hardware
//
#include <hip/hip_runtime.h>
#include <hip/hip_bf16.h>
#include <math.h>

typedef __attribute__((ext_vector_type(16))) _Float16 v16h;
typedef __attribute__((ext_vector_type(8)))  float    v8f;

#define DIMS 96
#define HEADS 2
#define CH 48
#define HID 255
#define BATCH 2
#define HW 256
#define SP (HW*HW)            // 65536 spatial per batch
#define NPIX (BATCH*SP)       // 131072 pixels
#define QKV_C 288
#define FFN_C 510
#define FFN_LD 512
#define GATE_LD 256

// A-fragment f16 16x32: lane L (row m = L&15, half hh = L>>4) holds elems e=0..15
// with K = (e<8) ? 8*hh+e : 16+8*hh+(e-8)
__device__ __forceinline__ int kmapA(int e, int hh) {
    return (e < 8) ? (8*hh + e) : (16 + 8*hh + (e - 8));
}

__device__ __forceinline__ v8f wmma_f16(v16h a, v16h b, v8f c) {
    return __builtin_amdgcn_wmma_f32_16x16x32_f16(false, a, false, b,
                                                  (short)0, c, false, false);
}

// ------------- f16 WMMA GEMM, 32x32 tile per wave (MT=2 x NT=2, 4 WMMA / K-step) -------------
// C[pix,n] = A16[pix, 0:K] x W16[K, :]  (+ residual if RES)
template<bool RES>
__global__ void gemm_f16_wmma(const _Float16* __restrict__ A,
                              const _Float16* __restrict__ W,
                              const float* __restrict__ residual,
                              float* __restrict__ C,
                              int K, int ldw, int ldc) {
    const int lane = threadIdx.x & 31;
    const int hh = lane >> 4, lo = lane & 15;
    const long p0 = (long)blockIdx.x * 32;     // two 16-row M tiles
    const int  n0 = blockIdx.y * 32;           // two 16-col N tiles
    v8f acc00 = {}, acc01 = {}, acc10 = {}, acc11 = {};
    for (int k0 = 0; k0 < K; k0 += 32) {
        const _Float16* ar0 = A + (p0 + lo) * (long)K + k0;
        const _Float16* ar1 = ar0 + 16 * (long)K;
        if (k0 + 32 < K) {                      // next A K-slab -> global_prefetch_b8
            __builtin_prefetch(ar0 + 32, 0, 3);
            __builtin_prefetch(ar1 + 32, 0, 3);
        }
        v16h a0, a1, b0, b1;
#pragma unroll
        for (int e = 0; e < 16; ++e) {
            int kk = 16*hh + e;
            a0[e] = ar0[kk];
            a1[e] = ar1[kk];
        }
        const _Float16* bc = W + (size_t)(k0 + 16*hh) * ldw + n0 + lo;
#pragma unroll
        for (int e = 0; e < 16; ++e) {
            size_t o = (size_t)e * ldw;
            b0[e] = bc[o];
            b1[e] = bc[o + 16];
        }
        acc00 = wmma_f16(a0, b0, acc00);
        acc01 = wmma_f16(a0, b1, acc01);
        acc10 = wmma_f16(a1, b0, acc10);
        acc11 = wmma_f16(a1, b1, acc11);
    }
#pragma unroll
    for (int r = 0; r < 8; ++r) {
        size_t i00 = (size_t)(p0 + r + 8*hh) * ldc + n0 + lo;       // M-tile 0, N-tile 0
        size_t i10 = i00 + (size_t)16 * ldc;                        // M-tile 1
        if (RES) {
            C[i00]      = acc00[r] + residual[i00];
            C[i00 + 16] = acc01[r] + residual[i00 + 16];
            C[i10]      = acc10[r] + residual[i10];
            C[i10 + 16] = acc11[r] + residual[i10 + 16];
        } else {
            C[i00]      = acc00[r];
            C[i00 + 16] = acc01[r];
            C[i10]      = acc10[r];
            C[i10 + 16] = acc11[r];
        }
    }
}

// ---------------- LayerNorm (over 96 ch) -> f16, one wave per pixel --------------------------
__global__ void ln_to_f16(const float* __restrict__ x, const float* __restrict__ gamma,
                          const float* __restrict__ beta, _Float16* __restrict__ out) {
    const int wave = threadIdx.x >> 5, lane = threadIdx.x & 31;
    const long pix = (long)blockIdx.x * 8 + wave;
    if (pix >= NPIX) return;
    const float* px = x + pix * DIMS;
    float v0 = px[lane], v1 = px[lane+32], v2 = px[lane+64];
    float s = v0+v1+v2, ss = v0*v0+v1*v1+v2*v2;
#pragma unroll
    for (int m = 16; m > 0; m >>= 1) { s += __shfl_xor(s, m, 32); ss += __shfl_xor(ss, m, 32); }
    float mu  = s * (1.0f/DIMS);
    float var = ss * (1.0f/DIMS) - mu*mu;
    float inv = rsqrtf(var + 1e-5f);
    _Float16* po = out + pix * DIMS;
    po[lane]    = (_Float16)((v0-mu)*inv*gamma[lane]    + beta[lane]);
    po[lane+32] = (_Float16)((v1-mu)*inv*gamma[lane+32] + beta[lane+32]);
    po[lane+64] = (_Float16)((v2-mu)*inv*gamma[lane+64] + beta[lane+64]);
}

// ---------------- depthwise 3x3 SAME (cross-correlation, NHWC) -------------------------------
__global__ void dwconv3x3(const float* __restrict__ in, const float* __restrict__ k9,
                          float* __restrict__ out, int C, int ld, long total) {
    long idx = (long)blockIdx.x * blockDim.x + threadIdx.x;
    if (idx >= total) return;
    int  c  = (int)(idx % C);
    long pix = idx / C;
    int ww = (int)(pix % HW); long t = pix / HW;
    int hh = (int)(t % HW);   int  b = (int)(t / HW);
    float acc = 0.f;
#pragma unroll
    for (int dy = -1; dy <= 1; ++dy)
#pragma unroll
        for (int dx = -1; dx <= 1; ++dx) {
            int yy = hh + dy, xx = ww + dx;
            if (yy < 0 || yy >= HW || xx < 0 || xx >= HW) continue;
            acc += in[(((long)b*HW + yy)*HW + xx)*ld + c] * k9[((dy+1)*3 + (dx+1))*C + c];
        }
    out[(((long)b*HW + hh)*HW + ww)*ld + c] = acc;
}

// ---------------- sum of squares over spatial for q,k channels (0..191 of qkv layout) --------
__global__ void sumsq_partial(const float* __restrict__ qkv, float* __restrict__ sums) {
    int b = blockIdx.x, grp = blockIdx.y, c = threadIdx.x;   // c in 0..191
    float acc = 0.f;
    long base = ((long)b*SP + (long)grp*256)*QKV_C + c;
    for (int i = 0; i < 256; ++i) { float v = qkv[base + (long)i*QKV_C]; acc += v*v; }
    atomicAdd(&sums[b*192 + c], acc);
}

__global__ void rsqrt_inplace(float* __restrict__ p, int n) {
    int i = blockIdx.x * blockDim.x + threadIdx.x;
    if (i < n) p[i] = rsqrtf(p[i]);
}

__global__ void fill_f32(float* __restrict__ p, long n) {
    long i = (long)blockIdx.x * blockDim.x + threadIdx.x;
    if (i < n) p[i] = 0.f;
}

// ---- attention logits: G[bh,c,d] += sum_s q[c,s]*k[d,s]; one q-tile vs all 3 d-tiles --------
__global__ void attn_logits_wmma(const float* __restrict__ qkv, float* __restrict__ G) {
    int id = blockIdx.x;            // bh*3 + ct
    int bh = id / 3, ct = id % 3;
    int b = bh >> 1, head = bh & 1;
    int lane = threadIdx.x & 31, hh = lane >> 4, lo = lane & 15;
    long sbase = (long)b*SP + (long)blockIdx.y * 1024;
    const float* qp = qkv + head*CH + ct*16 + lo;          // A rows: q channels (tile ct)
    const float* kp = qkv + DIMS + head*CH + lo;           // B cols: k channels (+16*dt)
    v8f acc[3] = {};
    for (int k0 = 0; k0 < 1024; k0 += 32) {
        v16h a;
#pragma unroll
        for (int e = 0; e < 16; ++e)
            a[e] = (_Float16)qp[(sbase + k0 + 16*hh + e) * QKV_C];
#pragma unroll
        for (int dt = 0; dt < 3; ++dt) {
            v16h bb;
#pragma unroll
            for (int e = 0; e < 16; ++e)
                bb[e] = (_Float16)kp[(sbase + k0 + 16*hh + e) * QKV_C + dt*16];
            acc[dt] = wmma_f16(a, bb, acc[dt]);
        }
    }
#pragma unroll
    for (int dt = 0; dt < 3; ++dt)
#pragma unroll
        for (int r = 0; r < 8; ++r)
            atomicAdd(&G[bh*2304 + (ct*16 + r + 8*hh)*CH + dt*16 + lo], acc[dt][r]);
}

// ---------------- scale by norms & temperature, softmax over d (48) --------------------------
__global__ void attn_softmax(float* __restrict__ G, const float* __restrict__ invn,
                             const float* __restrict__ temp) {
    int row = blockIdx.x;            // 0..191 = bh*48 + c
    int bh = row / CH, c = row % CH;
    int b = bh >> 1, head = bh & 1;
    int d = threadIdx.x;             // 64 threads, 48 active
    __shared__ float sm[64];
    float iq = invn[b*192 + head*CH + c];
    float tp = temp[head];
    float val = 0.f, v = -3.4e38f;
    if (d < CH) {
        val = G[bh*2304 + c*CH + d] * iq * invn[b*192 + DIMS + head*CH + d] * tp;
        v = val;
    }
    sm[d] = v; __syncthreads();
    for (int s = 32; s > 0; s >>= 1) { if (d < s) sm[d] = fmaxf(sm[d], sm[d+s]); __syncthreads(); }
    float mx = sm[0]; __syncthreads();
    float e = (d < CH) ? __expf(val - mx) : 0.f;
    sm[d] = e; __syncthreads();
    for (int s = 32; s > 0; s >>= 1) { if (d < s) sm[d] += sm[d+s]; __syncthreads(); }
    float inv = 1.f / sm[0];
    if (d < CH) G[bh*2304 + c*CH + d] = e * inv;
}

// ---------------- o = attn @ v (K=48 padded to 64), write f16 [pix,96] -----------------------
__global__ void attn_o_wmma(const float* __restrict__ P, const float* __restrict__ qkv,
                            _Float16* __restrict__ o16) {
    int id = blockIdx.x;             // bh*3 + mtile
    int bh = id / 3, mt = id % 3;
    int b = bh >> 1, head = bh & 1;
    int lane = threadIdx.x & 31, hh = lane >> 4, lo = lane & 15;
    long s0 = (long)b*SP + (long)blockIdx.y * 16;
    v8f acc = {};
#pragma unroll
    for (int k0 = 0; k0 < 64; k0 += 32) {
        v16h a, bb;
#pragma unroll
        for (int e = 0; e < 16; ++e) {
            int kk = k0 + 16*hh + e;
            a[e] = (kk < CH) ? (_Float16)P[bh*2304 + (mt*16 + lo)*CH + kk] : (_Float16)0.f;
        }
#pragma unroll
        for (int e = 0; e < 16; ++e) {
            int kk = k0 + 16*hh + e;
            bb[e] = (kk < CH) ? (_Float16)qkv[(s0 + lo)*QKV_C + 2*DIMS + head*CH + kk]
                              : (_Float16)0.f;
        }
        acc = wmma_f16(a, bb, acc);
    }
#pragma unroll
    for (int r = 0; r < 8; ++r)
        o16[(s0 + lo)*DIMS + head*CH + mt*16 + r + 8*hh] = (_Float16)acc[r];
}

// ---------------- gated exact GELU -> f16 [pix, 256] (col 255 = 0 pad) -----------------------
__global__ void gated_gelu(const float* __restrict__ hdd, _Float16* __restrict__ g16) {
    long i = (long)blockIdx.x * blockDim.x + threadIdx.x;
    if (i >= (long)NPIX * GATE_LD) return;
    int c = (int)(i % GATE_LD);
    long pix = i / GATE_LD;
    float g = 0.f;
    if (c < HID) {
        float x1 = hdd[pix*FFN_LD + c];
        float x2 = hdd[pix*FFN_LD + HID + c];
        g = 0.5f * x1 * (1.f + erff(x1 * 0.70710678118654752f)) * x2;
    }
    g16[i] = (_Float16)g;
}

// ---------------- fp32 -> f16 weight convert with zero padding -------------------------------
__global__ void cvt_pad_f16(const float* __restrict__ src, _Float16* __restrict__ dst,
                            int rs, int cs, int rd, int cd) {
    long i = (long)blockIdx.x * blockDim.x + threadIdx.x;
    if (i >= (long)rd * cd) return;
    int c = (int)(i % cd), r = (int)(i / cd);
    dst[i] = (r < rs && c < cs) ? (_Float16)src[(long)r*cs + c] : (_Float16)0.f;
}

// =============================================================================================
extern "C" void kernel_launch(void* const* d_in, const int* in_sizes, int n_in,
                              void* d_out, int out_size, void* d_ws, size_t ws_size,
                              hipStream_t stream) {
    (void)in_sizes; (void)n_in; (void)out_size; (void)ws_size;
    const float* x      = (const float*)d_in[0];
    const float* ln1_g  = (const float*)d_in[1];
    const float* ln1_b  = (const float*)d_in[2];
    const float* ln2_g  = (const float*)d_in[3];
    const float* ln2_b  = (const float*)d_in[4];
    const float* qkv_w  = (const float*)d_in[5];
    const float* qkv_dw = (const float*)d_in[6];
    const float* temp   = (const float*)d_in[7];
    const float* aout_w = (const float*)d_in[8];
    const float* fin_w  = (const float*)d_in[9];
    const float* ffn_dw = (const float*)d_in[10];
    const float* fout_w = (const float*)d_in[11];
    float* out = (float*)d_out;

    // ---- workspace carve-up ----
    char* base = (char*)d_ws;
    size_t off = 0;
    auto carve = [&](size_t bytes) { size_t o = off; off = (off + bytes + 255) & ~(size_t)255; return o; };
    _Float16* wq16  = (_Float16*)(base + carve((size_t)DIMS*QKV_C*2));
    _Float16* wo16  = (_Float16*)(base + carve((size_t)DIMS*DIMS*2));
    _Float16* wf1   = (_Float16*)(base + carve((size_t)DIMS*FFN_LD*2));
    _Float16* wf2   = (_Float16*)(base + carve((size_t)GATE_LD*DIMS*2));
    float*    norms = (float*)   (base + carve((size_t)BATCH*192*4));
    float*    G     = (float*)   (base + carve((size_t)BATCH*HEADS*CH*CH*4));
    _Float16* a16   = (_Float16*)(base + carve((size_t)NPIX*DIMS*2));
    _Float16* o16   = (_Float16*)(base + carve((size_t)NPIX*DIMS*2));
    _Float16* g16   = (_Float16*)(base + carve((size_t)NPIX*GATE_LD*2));
    float*    xmid  = (float*)   (base + carve((size_t)NPIX*DIMS*4));
    float*    t1    = (float*)   (base + carve((size_t)NPIX*FFN_LD*4));
    float*    t2    = (float*)   (base + carve((size_t)NPIX*FFN_LD*4));

    auto cdiv = [](long a, long b) { return (unsigned)((a + b - 1) / b); };

    // 0) weights -> f16 (with padding for FFN dims)
    cvt_pad_f16<<<cdiv((long)DIMS*QKV_C,256),256,0,stream>>>(qkv_w, wq16, DIMS, QKV_C, DIMS, QKV_C);
    cvt_pad_f16<<<cdiv((long)DIMS*DIMS,256),256,0,stream>>>(aout_w, wo16, DIMS, DIMS, DIMS, DIMS);
    cvt_pad_f16<<<cdiv((long)DIMS*FFN_LD,256),256,0,stream>>>(fin_w, wf1, DIMS, FFN_C, DIMS, FFN_LD);
    cvt_pad_f16<<<cdiv((long)GATE_LD*DIMS,256),256,0,stream>>>(fout_w, wf2, HID, DIMS, GATE_LD, DIMS);

    // 1) LN1 -> f16
    ln_to_f16<<<NPIX/8, 256, 0, stream>>>(x, ln1_g, ln1_b, a16);

    // 2) qkv GEMM [N,96]x[96,288] -> t1   (32x32 tiles: grid 4096 x 9)
    gemm_f16_wmma<false><<<dim3(NPIX/32, QKV_C/32), 32, 0, stream>>>(a16, wq16, nullptr, t1,
                                                                     DIMS, QKV_C, QKV_C);
    // 3) depthwise conv -> t2
    dwconv3x3<<<cdiv((long)NPIX*QKV_C,256),256,0,stream>>>(t1, qkv_dw, t2, QKV_C, QKV_C,
                                                           (long)NPIX*QKV_C);
    // 4) L2 norms of q,k over spatial
    fill_f32<<<cdiv(BATCH*192,256),256,0,stream>>>(norms, BATCH*192);
    fill_f32<<<cdiv((long)BATCH*HEADS*CH*CH,256),256,0,stream>>>(G, (long)BATCH*HEADS*CH*CH);
    sumsq_partial<<<dim3(BATCH, SP/256), 192, 0, stream>>>(t2, norms);
    rsqrt_inplace<<<cdiv(BATCH*192,256),256,0,stream>>>(norms, BATCH*192);

    // 5) logits (WMMA over spatial K, 3 d-tiles per wave), softmax
    attn_logits_wmma<<<dim3(BATCH*HEADS*3, SP/1024), 32, 0, stream>>>(t2, G);
    attn_softmax<<<BATCH*HEADS*CH, 64, 0, stream>>>(G, norms, temp);

    // 6) o = attn @ v -> f16
    attn_o_wmma<<<dim3(BATCH*HEADS*3, SP/16), 32, 0, stream>>>(G, t2, o16);

    // 7) x + o @ attn_out_w -> xmid   (grid 4096 x 3)
    gemm_f16_wmma<true><<<dim3(NPIX/32, DIMS/32), 32, 0, stream>>>(o16, wo16, x, xmid,
                                                                   DIMS, DIMS, DIMS);
    // 8) LN2 -> f16
    ln_to_f16<<<NPIX/8, 256, 0, stream>>>(xmid, ln2_g, ln2_b, a16);

    // 9) FFN-in GEMM [N,96]x[96,512pad] -> t2   (grid 4096 x 16)
    gemm_f16_wmma<false><<<dim3(NPIX/32, FFN_LD/32), 32, 0, stream>>>(a16, wf1, nullptr, t2,
                                                                      DIMS, FFN_LD, FFN_LD);
    // 10) depthwise conv (510 valid ch, stride 512) -> t1
    dwconv3x3<<<cdiv((long)NPIX*FFN_C,256),256,0,stream>>>(t2, ffn_dw, t1, FFN_C, FFN_LD,
                                                           (long)NPIX*FFN_C);
    // 11) gated exact GELU -> f16 [N,256pad]
    gated_gelu<<<cdiv((long)NPIX*GATE_LD,256),256,0,stream>>>(t1, g16);

    // 12) xmid + gated @ ffn_out_w -> out   (grid 4096 x 3)
    gemm_f16_wmma<true><<<dim3(NPIX/32, DIMS/32), 32, 0, stream>>>(g16, wf2, xmid, out,
                                                                   GATE_LD, DIMS, DIMS);
}
